// mHC_23905787970076
// MI455X (gfx1250) — matrix operand order repaired and checked by reading, not compiled, not run on hardware
//
#include <hip/hip_runtime.h>
#include <hip/hip_bf16.h>
#include <math.h>

// ---------------------------------------------------------------------------
// Types for CDNA5 WMMA (wave32)
// ---------------------------------------------------------------------------
typedef __attribute__((ext_vector_type(16))) __bf16 v16bf;
typedef __attribute__((ext_vector_type(8)))  __bf16 v8bf;
typedef __attribute__((ext_vector_type(4)))  __bf16 v4bf;
typedef __attribute__((ext_vector_type(8)))  float  v8f;

static __device__ inline v16bf cat8(v8bf lo, v8bf hi) {
    return __builtin_shufflevector(lo, hi, 0,1,2,3,4,5,6,7,8,9,10,11,12,13,14,15);
}

// Branch-free GELU (tanh approx):  tanh(u) = 1 - 2/(exp(2u)+1)
static __device__ inline float gelu_tanh(float v) {
    float u = 0.7978845608028654f * v * (1.0f + 0.044715f * v * v);
    float t = 1.0f - 2.0f / (__expf(2.0f * u) + 1.0f);
    return 0.5f * v * (1.0f + t);
}

// Problem constants
#define NB   4096      // batch
#define NS   4         // n streams
#define CD   2048      // d_model
#define FLT  (NS*CD)   // 8192
#define DFF  (4*CD)    // 8192

// GEMM tiling: 128x128 block tile, 4 waves (2x2), 64x64 per wave
#define BM 128
#define BN 128
#define BK 32
#define LDT 40         // padded LDS row pitch (bf16): 64B data + 16B pad = 80B
#define GEMM_THREADS 128

// ---------------------------------------------------------------------------
// Tensor Data Mover path (CDNA5): one DMA per 128x32 bf16 tile, LDS-padded
// to the 80B row pitch via the D# pad fields. This toolchain exposes the
// 6-argument builtin: (g0 u32x4, g1 i32x8, g2 i32x4, g3 i32x4, i32x8, cpol).
// ---------------------------------------------------------------------------
#if defined(__HIP_DEVICE_COMPILE__) && __has_builtin(__builtin_amdgcn_tensor_load_to_lds)
#define USE_TDM 1
typedef unsigned int u32x4 __attribute__((ext_vector_type(4)));
typedef int          i32x8 __attribute__((ext_vector_type(8)));
typedef int          i32x4 __attribute__((ext_vector_type(4)));

static __device__ inline void tdm_load_tile(const __bf16* gsrc, unsigned lds_off,
                                            unsigned k_stride) {
    unsigned long long ga = (unsigned long long)gsrc;
    u32x4 g0;
    g0[0] = 1u;                                        // count=1, user D#
    g0[1] = lds_off;                                   // lds_addr (bytes)
    g0[2] = (unsigned)ga;                              // global_addr[31:0]
    g0[3] = (unsigned)((ga >> 32) & 0x01FFFFFFu)       // global_addr[56:32]
          | (2u << 30);                                // type = 2 ("image")
    i32x8 g1;
    g1[0] = (1 << 16)                                  // data_size = 2 bytes
          | (1 << 20)                                  // pad_enable (load->LDS)
          | (3 << 22)                                  // pad_interval: 16 DWORDs (64B)
          | (3 << 25);                                 // pad_amount: 4 DWORDs (16B)
    g1[1] = (int)((k_stride & 0xFFFFu) << 16);         // tensor_dim0[15:0]
    g1[2] = (int)((k_stride >> 16) & 0xFFFFu)          // tensor_dim0[31:16]
          | (int)(8192u << 16);                        // tensor_dim1[15:0]
    g1[3] = (BK << 16);                                // tensor_dim1 hi | tile_dim0=32
    g1[4] = BM;                                        // tile_dim1=128, tile_dim2=0
    g1[5] = (int)k_stride;                             // tensor_dim0_stride[31:0]
    g1[6] = 0;                                         // stride hi | dim1_stride lo
    g1[7] = 0;
    i32x4 z4 = (i32x4){0, 0, 0, 0};                    // 2D tensor: groups 2/3 unused
    i32x8 z8 = (i32x8){0, 0, 0, 0, 0, 0, 0, 0};
    __builtin_amdgcn_tensor_load_to_lds(g0, g1, z4, z4, z8, 0);
}
#else
#define USE_TDM 0
#endif

// ---------------------------------------------------------------------------
// Kernel 0: tiled transpose + f32->bf16 convert.
//   src: (K, N) f32 row-major  ->  dst: (N, K) bf16 row-major (k-contiguous)
// ---------------------------------------------------------------------------
__global__ __launch_bounds__(256) void tcvt_kernel(const float* __restrict__ src,
                                                   __bf16* __restrict__ dst,
                                                   int K, int N) {
    __shared__ __bf16 tile[32 * 33];
    const int tid = threadIdx.x;
    const int k0 = blockIdx.x * 32;
    const int n0 = blockIdx.y * 32;
    {
        int kk = tid >> 3;
        int nn = (tid & 7) * 4;
        float4 v = *(const float4*)(src + (size_t)(k0 + kk) * N + n0 + nn);
        tile[(nn + 0) * 33 + kk] = (__bf16)v.x;
        tile[(nn + 1) * 33 + kk] = (__bf16)v.y;
        tile[(nn + 2) * 33 + kk] = (__bf16)v.z;
        tile[(nn + 3) * 33 + kk] = (__bf16)v.w;
    }
    __syncthreads();
    {
        int nn = tid >> 3;
        int kk = (tid & 7) * 4;
        v4bf o;
        o[0] = tile[nn * 33 + kk + 0];
        o[1] = tile[nn * 33 + kk + 1];
        o[2] = tile[nn * 33 + kk + 2];
        o[3] = tile[nn * 33 + kk + 3];
        *(v4bf*)(dst + (size_t)(n0 + nn) * K + k0 + kk) = o;
    }
}

// ---------------------------------------------------------------------------
// Kernel 1: gates + fin + residual.  One 256-thread block per batch row.
// ---------------------------------------------------------------------------
static __device__ inline float wave_red32(float v) {
    #pragma unroll
    for (int o = 16; o > 0; o >>= 1) v += __shfl_down(v, o, 32);
    return v;
}

__global__ __launch_bounds__(256) void gates_kernel(
    const float* __restrict__ x,
    const float* __restrict__ phi_pre, const float* __restrict__ phi_post,
    const float* __restrict__ phi_res,
    const float* __restrict__ bias_pre, const float* __restrict__ bias_post,
    const float* __restrict__ bias_res,
    const float* __restrict__ a_pre, const float* __restrict__ a_post,
    const float* __restrict__ a_res,
    float* __restrict__ out,           // (B, 4, 2048) residuals
    __bf16* __restrict__ finbf,        // (B, 2048)
    float* __restrict__ hpost_ws)      // (B, 4)
{
    __shared__ float xs[FLT];
    __shared__ float red[25 * 8];
    __shared__ float tot[25];
    __shared__ float g[20];

    const int b    = blockIdx.x;
    const int tid  = threadIdx.x;
    const int lane = tid & 31;
    const int w    = tid >> 5;
    const size_t xbase = (size_t)b * FLT;

    float acc[25];
    #pragma unroll
    for (int i = 0; i < 25; ++i) acc[i] = 0.0f;

    for (int f = tid; f < FLT; f += 256) {
        float xv = x[xbase + f];
        xs[f] = xv;
        acc[0] += xv * xv;
        float pp[4], po[4], pr[16];
        *(float4*)pp       = *(const float4*)(phi_pre  + (size_t)f * 4);
        *(float4*)po       = *(const float4*)(phi_post + (size_t)f * 4);
        *(float4*)(pr + 0) = *(const float4*)(phi_res  + (size_t)f * 16 + 0);
        *(float4*)(pr + 4) = *(const float4*)(phi_res  + (size_t)f * 16 + 4);
        *(float4*)(pr + 8) = *(const float4*)(phi_res  + (size_t)f * 16 + 8);
        *(float4*)(pr +12) = *(const float4*)(phi_res  + (size_t)f * 16 +12);
        #pragma unroll
        for (int j = 0; j < 4; ++j)  { acc[1 + j] += xv * pp[j]; acc[5 + j] += xv * po[j]; }
        #pragma unroll
        for (int j = 0; j < 16; ++j) { acc[9 + j] += xv * pr[j]; }
    }

    #pragma unroll
    for (int i = 0; i < 25; ++i) {
        float r = wave_red32(acc[i]);
        if (lane == 0) red[i * 8 + w] = r;
    }
    __syncthreads();
    if (tid < 25) {
        float s = 0.0f;
        #pragma unroll
        for (int j = 0; j < 8; ++j) s += red[tid * 8 + j];
        tot[tid] = s;
    }
    __syncthreads();

    if (tid == 0) {
        float inv_rms = rsqrtf(tot[0] * (1.0f / (float)FLT) + 1e-8f);
        float ap = a_pre[0], ao = a_post[0], ar = a_res[0];
        #pragma unroll
        for (int j = 0; j < 4; ++j) {
            float tpre  = ap * tot[1 + j] * inv_rms + bias_pre[j];
            g[j] = 1.0f / (1.0f + __expf(-tpre));
            float tpost = ao * tot[5 + j] * inv_rms + bias_post[j];
            hpost_ws[(size_t)b * 4 + j] = 2.0f / (1.0f + __expf(-tpost));
        }
        float M[16];
        #pragma unroll
        for (int i = 0; i < 16; ++i)
            M[i] = __expf(ar * tot[9 + i] * inv_rms + bias_res[i]);
        for (int it = 0; it < 20; ++it) {
            #pragma unroll
            for (int m = 0; m < 4; ++m) {
                float s = M[m*4+0] + M[m*4+1] + M[m*4+2] + M[m*4+3] + 1e-12f;
                float r = 1.0f / s;
                M[m*4+0] *= r; M[m*4+1] *= r; M[m*4+2] *= r; M[m*4+3] *= r;
            }
            #pragma unroll
            for (int n = 0; n < 4; ++n) {
                float s = M[n] + M[4+n] + M[8+n] + M[12+n] + 1e-12f;
                float r = 1.0f / s;
                M[n] *= r; M[4+n] *= r; M[8+n] *= r; M[12+n] *= r;
            }
        }
        #pragma unroll
        for (int i = 0; i < 16; ++i) g[4 + i] = M[i];
    }
    __syncthreads();

    float gp0 = g[0], gp1 = g[1], gp2 = g[2], gp3 = g[3];
    for (int c = tid; c < CD; c += 256) {
        float x0 = xs[c], x1 = xs[CD + c], x2 = xs[2*CD + c], x3 = xs[3*CD + c];
        float fin = gp0*x0 + gp1*x1 + gp2*x2 + gp3*x3;
        finbf[(size_t)b * CD + c] = (__bf16)fin;
        #pragma unroll
        for (int m = 0; m < 4; ++m) {
            float r = g[4 + m*4 + 0]*x0 + g[4 + m*4 + 1]*x1
                    + g[4 + m*4 + 2]*x2 + g[4 + m*4 + 3]*x3;
            out[(size_t)b * FLT + m * CD + c] = r;
        }
    }
}

// ---------------------------------------------------------------------------
// Shared GEMM mainloop: A (MxK) and Bt (NxK), both bf16 k-contiguous.
// 128x128 block tile, 4 waves, 64x64 per wave (16 WMMA per K-step per wave).
// LDS double-buffered, one barrier per K-step.
//   TDM path : wave0 issues TENSOR_LOAD_TO_LDS for next A/B tiles (padded),
//              s_wait_tensorcnt 0 + barrier at the bottom.
//   Fallback : register-staged global->LDS copies.
// ---------------------------------------------------------------------------
template<int KDIM>
static __device__ inline void gemm_body(const __bf16* __restrict__ A,
                                        const __bf16* __restrict__ Bt,
                                        __bf16 (&As)[2][BM * LDT],
                                        __bf16 (&Bs)[2][BM * LDT],
                                        v8f (&acc)[4][4]) {
    const int tid  = threadIdx.x;
    const int lane = tid & 31;
    const int w    = tid >> 5;          // 0..3
    const int wm   = (w >> 1) * 64;
    const int wn   = (w & 1) * 64;
    const int bm   = blockIdx.y * BM;
    const int bn   = blockIdx.x * BN;

    #pragma unroll
    for (int mi = 0; mi < 4; ++mi)
        #pragma unroll
        for (int ni = 0; ni < 4; ++ni)
            acc[mi][ni] = (v8f){0,0,0,0,0,0,0,0};

    const int arow  = wm + (lane & 15);
    const int koffA = (lane < 16) ? 0 : 8;
    const int bcol  = wn + (lane & 15);
    const int koffB = (lane < 16) ? 0 : 8;

#if USE_TDM
    const __bf16* Ag = A  + (size_t)bm * KDIM;
    const __bf16* Bg = Bt + (size_t)bn * KDIM;
    if (tid < 32) {
        tdm_load_tile(Ag, (unsigned)(unsigned long long)&As[0][0], KDIM);
        tdm_load_tile(Bg, (unsigned)(unsigned long long)&Bs[0][0], KDIM);
    }
    __builtin_amdgcn_s_wait_tensorcnt(0);
    __syncthreads();
#else
    const int r0  = tid >> 2;             // 0..31
    const int cg0 = (tid & 3) * 8;
    const __bf16* Ap = A  + (size_t)(bm + r0) * KDIM + cg0;
    const __bf16* Bp = Bt + (size_t)(bn + r0) * KDIM + cg0;
    v8bf ar[4], br[4];
    #pragma unroll
    for (int h = 0; h < 4; ++h) {
        ar[h] = *(const v8bf*)(Ap + (size_t)(32 * h) * KDIM);
        br[h] = *(const v8bf*)(Bp + (size_t)(32 * h) * KDIM);
    }
    #pragma unroll
    for (int h = 0; h < 4; ++h) {
        *(v8bf*)&As[0][(r0 + 32 * h) * LDT + cg0] = ar[h];
        *(v8bf*)&Bs[0][(r0 + 32 * h) * LDT + cg0] = br[h];
    }
    __syncthreads();
#endif

    for (int k0 = 0, buf = 0; k0 < KDIM; k0 += BK, buf ^= 1) {
        const int kn = k0 + BK;
#if USE_TDM
        if (kn < KDIM && tid < 32) {
            const int nb = buf ^ 1;
            tdm_load_tile(Ag + kn, (unsigned)(unsigned long long)&As[nb][0], KDIM);
            tdm_load_tile(Bg + kn, (unsigned)(unsigned long long)&Bs[nb][0], KDIM);
        }
#else
        if (kn < KDIM) {
            #pragma unroll
            for (int h = 0; h < 4; ++h) {
                ar[h] = *(const v8bf*)(Ap + (size_t)(32 * h) * KDIM + kn);
                br[h] = *(const v8bf*)(Bp + (size_t)(32 * h) * KDIM + kn);
            }
        }
        if (k0 + 2 * BK < KDIM) {
            __builtin_prefetch(Ap + k0 + 2 * BK, 0, 1);
            __builtin_prefetch(Bp + k0 + 2 * BK, 0, 1);
        }
#endif
        const __bf16* as = As[buf];
        const __bf16* bs = Bs[buf];
        v16bf af[4], bf[4];
        #pragma unroll
        for (int mi = 0; mi < 4; ++mi) {
            v8bf lo = *(const v8bf*)&as[(arow + mi * 16) * LDT + koffA];
            v8bf hi = *(const v8bf*)&as[(arow + mi * 16) * LDT + 16 + koffA];
            af[mi] = cat8(lo, hi);
        }
        #pragma unroll
        for (int ni = 0; ni < 4; ++ni) {
            v8bf lo = *(const v8bf*)&bs[(bcol + ni * 16) * LDT + koffB];
            v8bf hi = *(const v8bf*)&bs[(bcol + ni * 16) * LDT + 16 + koffB];
            bf[ni] = cat8(lo, hi);
        }
        #pragma unroll
        for (int mi = 0; mi < 4; ++mi)
            #pragma unroll
            for (int ni = 0; ni < 4; ++ni)
                acc[mi][ni] = __builtin_amdgcn_wmma_f32_16x16x32_bf16(
                    false, af[mi], false, bf[ni], (short)0, acc[mi][ni],
                    false, false);
#if USE_TDM
        __builtin_amdgcn_s_wait_tensorcnt(0);
#else
        if (kn < KDIM) {
            const int nb = buf ^ 1;
            #pragma unroll
            for (int h = 0; h < 4; ++h) {
                *(v8bf*)&As[nb][(r0 + 32 * h) * LDT + cg0] = ar[h];
                *(v8bf*)&Bs[nb][(r0 + 32 * h) * LDT + cg0] = br[h];
            }
        }
#endif
        __syncthreads();
    }
}

// ---------------------------------------------------------------------------
// Kernel 2: GEMM1   Hidden = gelu(Fin @ W1 + b1)
// ---------------------------------------------------------------------------
__global__ __launch_bounds__(GEMM_THREADS) void gemm1_kernel(
    const __bf16* __restrict__ A,    // Fin bf16 (4096, 2048)
    const __bf16* __restrict__ Bt,   // W1^T bf16 (8192, 2048)
    const float*  __restrict__ b1,
    __bf16* __restrict__ H)          // (4096, 8192)
{
    __shared__ __align__(16) __bf16 As[2][BM * LDT];
    __shared__ __align__(16) __bf16 Bs[2][BM * LDT];
    v8f acc[4][4];
    gemm_body<CD>(A, Bt, As, Bs, acc);

    const int tid  = threadIdx.x;
    const int lane = tid & 31;
    const int w    = tid >> 5;
    const int wm   = (w >> 1) * 64;
    const int wn   = (w & 1) * 64;
    const int bm   = blockIdx.y * BM;
    const int bn   = blockIdx.x * BN;

    #pragma unroll
    for (int ni = 0; ni < 4; ++ni) {
        int n = bn + wn + ni * 16 + (lane & 15);
        float bias = b1[n];
        #pragma unroll
        for (int mi = 0; mi < 4; ++mi) {
            int mbase = bm + wm + mi * 16 + ((lane >> 4) << 3);
            #pragma unroll
            for (int r = 0; r < 8; ++r) {
                float v = acc[mi][ni][r] + bias;
                H[(size_t)(mbase + r) * DFF + n] = (__bf16)gelu_tanh(v);
            }
        }
    }
}

// ---------------------------------------------------------------------------
// Kernel 3: GEMM2   fout = Hidden @ W2 + b2; out += H_post[m]*fout
// ---------------------------------------------------------------------------
__global__ __launch_bounds__(GEMM_THREADS) void gemm2_kernel(
    const __bf16* __restrict__ A,    // Hidden bf16 (4096, 8192)
    const __bf16* __restrict__ Bt,   // W2^T bf16 (2048, 8192)
    const float*  __restrict__ b2,
    const float*  __restrict__ hpost,// (4096, 4)
    float* __restrict__ out)         // (4096, 4, 2048) holds residuals
{
    __shared__ __align__(16) __bf16 As[2][BM * LDT];
    __shared__ __align__(16) __bf16 Bs[2][BM * LDT];
    v8f acc[4][4];
    gemm_body<DFF>(A, Bt, As, Bs, acc);

    const int tid  = threadIdx.x;
    const int lane = tid & 31;
    const int w    = tid >> 5;
    const int wm   = (w >> 1) * 64;
    const int wn   = (w & 1) * 64;
    const int bm   = blockIdx.y * BM;
    const int bn   = blockIdx.x * BN;

    #pragma unroll
    for (int ni = 0; ni < 4; ++ni) {
        int n = bn + wn + ni * 16 + (lane & 15);
        float bias = b2[n];
        #pragma unroll
        for (int mi = 0; mi < 4; ++mi) {
            int mbase = bm + wm + mi * 16 + ((lane >> 4) << 3);
            #pragma unroll
            for (int r = 0; r < 8; ++r) {
                int bidx = mbase + r;
                float fout = acc[mi][ni][r] + bias;
                float hp[4];
                *(float4*)hp = *(const float4*)(hpost + (size_t)bidx * 4);
                size_t ob = (size_t)bidx * FLT + n;
                #pragma unroll
                for (int ms = 0; ms < 4; ++ms)
                    out[ob + (size_t)ms * CD] += hp[ms] * fout;
            }
        }
    }
}

// ---------------------------------------------------------------------------
// Host launcher
// ---------------------------------------------------------------------------
extern "C" void kernel_launch(void* const* d_in, const int* in_sizes, int n_in,
                              void* d_out, int out_size, void* d_ws, size_t ws_size,
                              hipStream_t stream) {
    const float* x        = (const float*)d_in[0];
    const float* phi_pre  = (const float*)d_in[1];
    const float* phi_post = (const float*)d_in[2];
    const float* phi_res  = (const float*)d_in[3];
    const float* bias_pre = (const float*)d_in[4];
    const float* bias_post= (const float*)d_in[5];
    const float* bias_res = (const float*)d_in[6];
    const float* a_pre    = (const float*)d_in[7];
    const float* a_post   = (const float*)d_in[8];
    const float* a_res    = (const float*)d_in[9];
    const float* W1       = (const float*)d_in[10];
    const float* b1       = (const float*)d_in[11];
    const float* W2       = (const float*)d_in[12];
    const float* b2       = (const float*)d_in[13];
    float* out            = (float*)d_out;

    char* ws = (char*)d_ws;
    const size_t OFF_FIN   = 0;                                   // 16 MB
    const size_t OFF_HPOST = OFF_FIN   + (size_t)NB * CD * 2;     // 64 KB
    const size_t OFF_W1T   = OFF_HPOST + (size_t)NB * 4  * 4;     // 32 MB
    const size_t OFF_W2T   = OFF_W1T   + (size_t)CD * DFF * 2;    // 32 MB
    const size_t OFF_HID   = OFF_W2T   + (size_t)DFF * CD * 2;    // 64 MB

    __bf16* finbf = (__bf16*)(ws + OFF_FIN);
    float*  hpost = (float*) (ws + OFF_HPOST);
    __bf16* w1t   = (__bf16*)(ws + OFF_W1T);   // (8192, 2048)
    __bf16* w2t   = (__bf16*)(ws + OFF_W2T);   // (2048, 8192)
    __bf16* hid   = (__bf16*)(ws + OFF_HID);   // (4096, 8192)

    tcvt_kernel<<<dim3(CD / 32,  DFF / 32), 256, 0, stream>>>(W1, w1t, CD,  DFF);
    tcvt_kernel<<<dim3(DFF / 32, CD / 32),  256, 0, stream>>>(W2, w2t, DFF, CD);

    gates_kernel<<<NB, 256, 0, stream>>>(x, phi_pre, phi_post, phi_res,
                                         bias_pre, bias_post, bias_res,
                                         a_pre, a_post, a_res,
                                         out, finbf, hpost);

    gemm1_kernel<<<dim3(DFF / BN, NB / BM), GEMM_THREADS, 0, stream>>>(finbf, w1t, b1, hid);
    gemm2_kernel<<<dim3(CD / BN, NB / BM), GEMM_THREADS, 0, stream>>>(hid, w2t, b2, hpost, out);
}
